// GNN_652835029171
// MI455X (gfx1250) — matrix operand order repaired and checked by reading, not compiled, not run on hardware
//
#include <hip/hip_runtime.h>

// GAT (2-layer) for MI455X / gfx1250: WMMA f16->f32 GEMM + atomic edge softmax.
// N=50000, D=128, E=625000 (+N self loops). All fp32 I/O; f16 inside WMMA.

constexpr int   HD        = 128;
constexpr float NEG_SLOPE = 0.2f;

typedef __attribute__((ext_vector_type(16))) _Float16 v16h;
typedef __attribute__((ext_vector_type(8)))  float    v8f;

// ---------------- K0: Wt[n][k] = (f16) W[k][n] ----------------
__global__ void k_transpose_w(const float* __restrict__ W, _Float16* __restrict__ Wt) {
    int k = blockIdx.x * 16 + threadIdx.x;
    int n = blockIdx.y * 16 + threadIdx.y;
    Wt[n * HD + k] = (_Float16)W[k * HD + n];
}

// ---------------- K1: h = (relu?) x @ W  via v_wmma_f32_16x16x32_f16 ----------------
// Block = 256 threads = 8 wave32; block handles a 16-row tile, wave w -> cols [16w,16w+16).
// A layout (16-bit 16x32): lane<16 -> M=lane, K = kb..kb+7 (v0..v3 pairs) and kb+16..kb+23
//                          lane>=16 -> same M, kb shifted by +8.
// B layout (16-bit 32x16): lane&15 = N, K same as A (kb..kb+7, kb+16..kb+23; kb +8 for lane>=16).
__global__ void k_gemm_wmma(const float* __restrict__ x, const _Float16* __restrict__ Wt,
                            float* __restrict__ h, int apply_relu, int nrows) {
    const int lane = threadIdx.x & 31;
    const int wave = threadIdx.x >> 5;
    const int row0 = blockIdx.x * 16;
    const int col0 = wave * 16;
    const int r    = lane & 15;
    const int hi   = lane >> 4;

    const float*    xrow = x  + (size_t)(row0 + r) * HD;
    const _Float16* wcol = Wt + (size_t)(lane & 15) * HD;

    // Prefetch next row tile of x (speculative; dropped if OOB).
    __builtin_prefetch(xrow + 16 * HD, 0, 1);

    v8f c = {};
#pragma unroll
    for (int k0 = 0; k0 < HD; k0 += 32) {
        const int kb = k0 + hi * 8;
        v16h a;
#pragma unroll
        for (int i = 0; i < 8; ++i) {
            float v0 = xrow[kb + i];
            float v1 = xrow[kb + 16 + i];
            if (apply_relu) { v0 = fmaxf(v0, 0.0f); v1 = fmaxf(v1, 0.0f); }
            a[i]     = (_Float16)v0;
            a[8 + i] = (_Float16)v1;
        }
        const int kb2 = k0 + hi * 8;
        v16h b;
#pragma unroll
        for (int i = 0; i < 8; ++i) { b[i] = wcol[kb2 + i]; b[8 + i] = wcol[kb2 + 16 + i]; }

        // 8 args: (neg_a, A, neg_b, B, c_mod, C, reuse_a, reuse_b)
        c = __builtin_amdgcn_wmma_f32_16x16x32_f16(false, a, false, b, (short)0, c,
                                                   false, false);
    }
    // D layout: VGPR rr holds (M = rr + 8*hi, N = lane&15)
    float* hp = h + (size_t)(row0 + hi * 8) * HD + col0 + (lane & 15);
#pragma unroll
    for (int rr = 0; rr < 8; ++rr) hp[(size_t)rr * HD] = c[rr];
    (void)nrows;
}

// ---------------- K2: per-row scores + init m/z (wave per row) ----------------
__global__ void k_scores(const float* __restrict__ h, const float* __restrict__ a_src,
                         const float* __restrict__ a_dst, float* __restrict__ s_src,
                         float* __restrict__ s_dst, float* __restrict__ m,
                         float* __restrict__ z, int N) {
    const int wave = threadIdx.x >> 5;
    const int lane = threadIdx.x & 31;
    const int row  = blockIdx.x * 8 + wave;
    if (row >= N) return;
    const float4 hx = ((const float4*)(h + (size_t)row * HD))[lane];
    const float4 av = ((const float4*)a_src)[lane];
    const float4 dv = ((const float4*)a_dst)[lane];
    float ss = hx.x * av.x + hx.y * av.y + hx.z * av.z + hx.w * av.w;
    float sd = hx.x * dv.x + hx.y * dv.y + hx.z * dv.z + hx.w * dv.w;
#pragma unroll
    for (int off = 16; off > 0; off >>= 1) {
        ss += __shfl_xor(ss, off, 32);
        sd += __shfl_xor(sd, off, 32);
    }
    if (lane == 0) {
        s_src[row] = ss;
        s_dst[row] = sd;
        m[row]     = -__builtin_inff();
        z[row]     = 0.0f;
    }
}

// Float atomic max via signed/unsigned integer monotone mapping.
__device__ __forceinline__ void atomicMaxF(float* addr, float v) {
    if (v >= 0.0f) atomicMax((int*)addr, __float_as_int(v));
    else           atomicMin((unsigned int*)addr, __float_as_uint(v));
}

// ---------------- K3: edge logits + segment max ----------------
__global__ void k_edge_logit(const int* __restrict__ ei, const float* __restrict__ s_src,
                             const float* __restrict__ s_dst, float* __restrict__ pbuf,
                             float* __restrict__ m, int E, int Etot) {
    int e = blockIdx.x * blockDim.x + threadIdx.x;
    if (e >= Etot) return;
    const int src = (e < E) ? ei[e]     : (e - E);
    const int dst = (e < E) ? ei[E + e] : (e - E);
    float v = s_src[src] + s_dst[dst];
    v = (v > 0.0f) ? v : NEG_SLOPE * v;
    pbuf[e] = v;
    atomicMaxF(m + dst, v);
}

// ---------------- K4: p = exp(e - m[dst]); z[dst] += p ----------------
__global__ void k_edge_exp(const int* __restrict__ ei, const float* __restrict__ m,
                           float* __restrict__ pbuf, float* __restrict__ z,
                           int E, int Etot) {
    int e = blockIdx.x * blockDim.x + threadIdx.x;
    if (e >= Etot) return;
    const int dst = (e < E) ? ei[E + e] : (e - E);
    const float p = __expf(pbuf[e] - m[dst]);
    pbuf[e] = p;
    atomicAdd(z + dst, p);
}

// ---------------- K5: out[i] = bias[i%128] ----------------
__global__ void k_init_out(const float* __restrict__ b, float* __restrict__ out,
                           long long total) {
    long long i = (long long)blockIdx.x * blockDim.x + threadIdx.x;
    if (i < total) out[i] = b[i & (HD - 1)];
}

// ---------------- K6: out[dst] += alpha * h[src]  (wave per edge) ----------------
__global__ void k_edge_aggregate(const int* __restrict__ ei, const float* __restrict__ pbuf,
                                 const float* __restrict__ z, const float* __restrict__ h,
                                 float* __restrict__ out, int E, int Etot) {
    const long long gid = (long long)blockIdx.x * blockDim.x + threadIdx.x;
    const int e    = (int)(gid >> 5);
    const int lane = (int)(gid & 31);
    if (e >= Etot) return;
    const int src = (e < E) ? ei[e]     : (e - E);
    const int dst = (e < E) ? ei[E + e] : (e - E);
    const float alpha = pbuf[e] / (z[dst] + 1e-16f);
    const float4 hx = ((const float4*)(h + (size_t)src * HD))[lane];  // b128 coalesced
    float* op = out + (size_t)dst * HD + lane * 4;
    atomicAdd(op + 0, alpha * hx.x);
    atomicAdd(op + 1, alpha * hx.y);
    atomicAdd(op + 2, alpha * hx.z);
    atomicAdd(op + 3, alpha * hx.w);
}

extern "C" void kernel_launch(void* const* d_in, const int* in_sizes, int n_in,
                              void* d_out, int out_size, void* d_ws, size_t ws_size,
                              hipStream_t stream) {
    const float* x   = (const float*)d_in[0];
    const int*   ei  = (const int*)d_in[1];   // [2,E] flat: row0=src, row1=dst
    const float* W1  = (const float*)d_in[2];
    const float* as1 = (const float*)d_in[3];
    const float* ad1 = (const float*)d_in[4];
    const float* b1  = (const float*)d_in[5];
    const float* W2  = (const float*)d_in[6];
    const float* as2 = (const float*)d_in[7];
    const float* ad2 = (const float*)d_in[8];
    const float* b2  = (const float*)d_in[9];
    float* outF = (float*)d_out;

    const int N    = in_sizes[0] / HD;   // 50000 (multiple of 16)
    const int E    = in_sizes[1] / 2;    // 625000
    const int Etot = E + N;              // self loops appended

    // ---- workspace layout ----
    char*  ws  = (char*)d_ws;
    size_t off = 0;
    auto alloc = [&](size_t bytes) -> void* {
        void* p = ws + off;
        off += (bytes + 255) & ~(size_t)255;
        return p;
    };
    _Float16* Wt   = (_Float16*)alloc((size_t)HD * HD * sizeof(_Float16));
    float*    hbuf = (float*)alloc((size_t)N * HD * sizeof(float));
    float*    out1 = (float*)alloc((size_t)N * HD * sizeof(float));
    float*    ssrc = (float*)alloc((size_t)N * sizeof(float));
    float*    sdst = (float*)alloc((size_t)N * sizeof(float));
    float*    mbuf = (float*)alloc((size_t)N * sizeof(float));
    float*    zbuf = (float*)alloc((size_t)N * sizeof(float));
    float*    pbuf = (float*)alloc((size_t)Etot * sizeof(float));
    (void)ws_size; (void)n_in; (void)out_size;

    const int gGemm  = N / 16;
    const int gRows  = (N + 7) / 8;
    const int gEdge  = (Etot + 255) / 256;
    const int gEdgeW = (int)(((long long)Etot * 32 + 255) / 256);
    const int gInit  = (int)(((long long)N * HD + 255) / 256);

    auto run_layer = [&](const float* xin, int relu_in, const float* W,
                         const float* a_s, const float* a_d, const float* bias,
                         float* outp) {
        k_transpose_w<<<dim3(8, 8), dim3(16, 16), 0, stream>>>(W, Wt);
        k_gemm_wmma<<<gGemm, 256, 0, stream>>>(xin, Wt, hbuf, relu_in, N);
        k_scores<<<gRows, 256, 0, stream>>>(hbuf, a_s, a_d, ssrc, sdst, mbuf, zbuf, N);
        k_edge_logit<<<gEdge, 256, 0, stream>>>(ei, ssrc, sdst, pbuf, mbuf, E, Etot);
        k_edge_exp<<<gEdge, 256, 0, stream>>>(ei, mbuf, pbuf, zbuf, E, Etot);
        k_init_out<<<gInit, 256, 0, stream>>>(bias, outp, (long long)N * HD);
        k_edge_aggregate<<<gEdgeW, 256, 0, stream>>>(ei, pbuf, zbuf, hbuf, outp, E, Etot);
    };

    // Layer 1: plain x in, result -> out1 (ReLU deferred: fused into layer-2 A loads)
    run_layer(x, 0, W1, as1, ad1, b1, out1);
    // Layer 2: relu(out1) in, result -> d_out
    run_layer(out1, 1, W2, as2, ad2, b2, outF);
}
